// SwinIRBlock_60378650247692
// MI455X (gfx1250) — matrix operand order repaired and checked by reading, not compiled, not run on hardware
//
#include <hip/hip_runtime.h>
#include <math.h>

// ---------------- types & helpers ----------------
typedef __attribute__((ext_vector_type(16))) __bf16 v16bf;
typedef __attribute__((ext_vector_type(8)))  float  v8f;
typedef __attribute__((ext_vector_type(4)))  unsigned u32x4;

union Frag {
    v16bf v;
    unsigned u[8];
    unsigned short s[16];
    u32x4 q[2];
};

__device__ __forceinline__ unsigned short f2bf(float f) {
    unsigned x = __float_as_uint(f);
    unsigned r = (x + 0x7FFFu + ((x >> 16) & 1u)) >> 16;
    return (unsigned short)r;
}

// A-fragment (16x32 bf16, one k-step): lane row = lane%16,
//   elements 0..15 -> k = hi16+0..15  (hi16 = lane>=16 ? 16:0)
// Both halves are 16B-contiguous and 16B-aligned for our strides.
__device__ __forceinline__ void loadA16(Frag& f, const unsigned short* base) {
    f.q[0] = *(const u32x4*)(base);
    f.q[1] = *(const u32x4*)(base + 8);
}
// B-fragment (32x16 bf16): lane col = lane%16, elements 0..15 -> k = hi16+0..15
__device__ __forceinline__ void loadB16(Frag& f, const unsigned short* base) {
    f.q[0] = *(const u32x4*)(base);
    f.q[1] = *(const u32x4*)(base + 8);
}

// Constants for this problem
#define BB   4
#define CC   192
#define HH   256
#define WW2  256
#define NHD  6
#define HD   32
#define NTOK 64            // tokens per window
#define NWIN 1024          // windows per batch image (32x32)
#define MTOK (BB * HH * WW2)   // 262144 total tokens

// ---------------- fp32 -> bf16 convert ----------------
__global__ void cvt_bf16(const float* __restrict__ src,
                         unsigned short* __restrict__ dst, int n) {
    int i = blockIdx.x * blockDim.x + threadIdx.x;
    if (i < n) dst[i] = f2bf(src[i]);
}

// ---------------- shifted-window gather (NCHW f32 -> windowed bf16) --------
__global__ __launch_bounds__(256) void gather_windows(
    const float* __restrict__ x, const float* __restrict__ feat,
    unsigned short* __restrict__ Xw, unsigned short* __restrict__ Fw) {
    long t = (long)blockIdx.x * blockDim.x + threadIdx.x;
    const long tot = (long)BB * CC * HH * WW2;
    if (t >= tot) return;
    int w  = (int)(t & 255);
    int h  = (int)((t >> 8) & 255);
    int bc = (int)(t >> 16);
    int c  = bc % CC;
    int b  = bc / CC;
    int h2 = (h - 4) & 255;       // roll(-4): this pixel lands at shifted (h2,w2)
    int w2 = (w - 4) & 255;
    int win = b * NWIN + (h2 >> 3) * 32 + (w2 >> 3);
    int n   = ((h2 & 7) << 3) + (w2 & 7);
    long dst = ((long)win * NTOK + n) * CC + c;
    Xw[dst] = f2bf(x[t]);
    Fw[dst] = f2bf(feat[t]);
}

// ---------------- generic bf16 WMMA GEMM: D = act((A*W^T + bias)*alpha) ----
// A: [M x K] bf16 row-major; W: [O x K] bf16 row-major; D: [M x O]
// One wave computes a 16x64 tile of D. ACT: 0 none, 1 exact GELU.
template <int ACT, int OUTF32>
__global__ __launch_bounds__(256) void gemm_bf16(
    const unsigned short* __restrict__ A,
    const unsigned short* __restrict__ Wt,
    const float* __restrict__ bias,   // may be null
    void* __restrict__ Dst,
    int M, int K, int O, float alpha) {
    const int lane = threadIdx.x & 31;
    const int widx = threadIdx.x >> 5;
    const long tile = (long)blockIdx.x * 8 + widx;
    const int oTiles = O >> 6;
    const long mt = tile / oTiles;
    const int ot = (int)(tile % oTiles);
    if (mt >= (M >> 4)) return;
    const long m0 = mt << 4;
    const int o0 = ot << 6;
    const int lr = lane & 15;
    const int hi8  = (lane & 16) ? 8 : 0;
    const int hi16 = (lane & 16) ? 16 : 0;

    v8f acc[4] = {v8f{}, v8f{}, v8f{}, v8f{}};
    const unsigned short* Arow = A + (m0 + lr) * (long)K + hi16;
    const unsigned short* W0 = Wt + (long)(o0 + lr) * K + hi16;
    for (int k0 = 0; k0 < K; k0 += 32) {
        if (k0 + 64 < K) __builtin_prefetch(Arow + k0 + 64, 0, 0);
        Frag a;
        loadA16(a, Arow + k0);
#pragma unroll
        for (int t = 0; t < 4; ++t) {
            Frag b;
            loadB16(b, W0 + (long)t * 16 * K + k0);
            acc[t] = __builtin_amdgcn_wmma_f32_16x16x32_bf16(
                false, a.v, false, b.v, (short)0, acc[t], false, false);
        }
    }
#pragma unroll
    for (int t = 0; t < 4; ++t) {
        int oc = o0 + t * 16 + lr;
        float bv = bias ? bias[oc] : 0.0f;
#pragma unroll
        for (int r = 0; r < 8; ++r) {
            long mr = m0 + r + hi8;                           // D layout
            float v = (acc[t][r] + bv) * alpha;
            if (ACT == 1) v = 0.5f * v * (1.0f + erff(v * 0.70710678118f));
            if (OUTF32) ((float*)Dst)[mr * O + oc] = v;
            else        ((unsigned short*)Dst)[mr * O + oc] = f2bf(v);
        }
    }
}

// ---------------- windowed attention: one wave per (window, head) ----------
__global__ __launch_bounds__(256) void attn_kernel(
    const unsigned short* __restrict__ Q,    // [MTOK x 192] (pre-scaled)
    const unsigned short* __restrict__ KV,   // [MTOK x 384] (k cols 0..191, v 192..383)
    const float* __restrict__ mask,          // [1024 x 64 x 64]
    const int* __restrict__ relIdx,          // [64 x 64]
    const float* __restrict__ biasTab,       // [225 x 6]
    unsigned short* __restrict__ Out) {      // [MTOK x 192]
    __shared__ unsigned short attS[8 * NTOK * NTOK];  // 64KB, per-wave 8KB slabs
    const int lane = threadIdx.x & 31;
    const int widx = threadIdx.x >> 5;
    const int job = blockIdx.x * 8 + widx;
    if (job >= BB * NWIN * NHD) return;
    const int win = job / NHD;
    const int head = job % NHD;
    const int wm = win & (NWIN - 1);
    const int lr = lane & 15;
    const int hi8 = (lane & 16) ? 8 : 0;
    const int hi16 = (lane & 16) ? 16 : 0;
    unsigned short* myAtt = attS + widx * (NTOK * NTOK);
    const long tok0 = (long)win * NTOK;

    // Preload V B-fragments (kc = n-chunk, dt = d-tile); col = lane%16, k = token
    Frag vf[2][2];
#pragma unroll
    for (int kc = 0; kc < 2; ++kc)
#pragma unroll
        for (int dt = 0; dt < 2; ++dt)
#pragma unroll
            for (int e = 0; e < 16; ++e) {
                int k = e + hi16 + kc * 32;
                vf[kc][dt].s[e] =
                    KV[(tok0 + k) * 384 + 192 + head * HD + dt * 16 + lr];
            }

    // S = Q K^T (K-dim = hd = 32 -> one WMMA per 16x16 tile), then softmax row-wise
    for (int mtI = 0; mtI < 4; ++mtI) {
        Frag aq;
        loadA16(aq, Q + (tok0 + mtI * 16 + lr) * CC + head * HD + hi16);
        v8f s[4];
#pragma unroll
        for (int nt = 0; nt < 4; ++nt) {
            Frag bk;
            loadB16(bk, KV + (tok0 + nt * 16 + lr) * 384 + head * HD + hi16);
            v8f z = {};
            s[nt] = __builtin_amdgcn_wmma_f32_16x16x32_bf16(
                false, aq.v, false, bk.v, (short)0, z, false, false);
        }
        // rel-bias + shift-mask + softmax (row m spans one 16-lane half-wave)
#pragma unroll
        for (int r = 0; r < 8; ++r) {
            int m = mtI * 16 + r + hi8;
            float vals[4];
#pragma unroll
            for (int nt = 0; nt < 4; ++nt) {
                int n = nt * 16 + lr;
                float bv = biasTab[relIdx[m * NTOK + n] * NHD + head];
                vals[nt] = s[nt][r] + bv + mask[((long)wm * NTOK + m) * NTOK + n];
            }
            float mx = fmaxf(fmaxf(vals[0], vals[1]), fmaxf(vals[2], vals[3]));
            for (int d = 8; d >= 1; d >>= 1) mx = fmaxf(mx, __shfl_xor(mx, d, 32));
            float sum = 0.f;
#pragma unroll
            for (int nt = 0; nt < 4; ++nt) {
                vals[nt] = __expf(vals[nt] - mx);
                sum += vals[nt];
            }
            for (int d = 8; d >= 1; d >>= 1) sum += __shfl_xor(sum, d, 32);
            float inv = 1.0f / sum;
#pragma unroll
            for (int nt = 0; nt < 4; ++nt)
                myAtt[m * NTOK + nt * 16 + lr] = f2bf(vals[nt] * inv);
        }
    }

    // P = att @ V  (DS ops from the same wave are in-order; per-wave LDS slab)
    for (int mtI = 0; mtI < 4; ++mtI) {
        v8f p[2] = {v8f{}, v8f{}};
#pragma unroll
        for (int kc = 0; kc < 2; ++kc) {
            Frag aa;
            loadA16(aa, myAtt + (mtI * 16 + lr) * NTOK + kc * 32 + hi16);
#pragma unroll
            for (int dt = 0; dt < 2; ++dt)
                p[dt] = __builtin_amdgcn_wmma_f32_16x16x32_bf16(
                    false, aa.v, false, vf[kc][dt].v, (short)0, p[dt], false, false);
        }
#pragma unroll
        for (int dt = 0; dt < 2; ++dt)
#pragma unroll
            for (int r = 0; r < 8; ++r)
                Out[(tok0 + mtI * 16 + r + hi8) * CC + head * HD + dt * 16 + lr] =
                    f2bf(p[dt][r]);
    }
}

// ---------------- merge-LN + shortcut (window order -> NHWC) ---------------
__global__ __launch_bounds__(256) void merge_ln_add(
    const float* __restrict__ MM,   // [MTOK x 192] window-token order
    const float* __restrict__ x,    // NCHW f32 (shortcut, un-shifted)
    const float* __restrict__ g, const float* __restrict__ bt,
    float* __restrict__ X1, unsigned short* __restrict__ X1b) {
    const int lane = threadIdx.x & 31;
    const int widx = threadIdx.x >> 5;
    long p = (long)blockIdx.x * 8 + widx;          // NHWC pixel row
    if (p >= (long)MTOK) return;
    int w = (int)(p & 255);
    int h = (int)((p >> 8) & 255);
    int b = (int)(p >> 16);
    int h2 = (h - 4) & 255;
    int w2 = (w - 4) & 255;
    long mmrow = (long)(b * NWIN + (h2 >> 3) * 32 + (w2 >> 3)) * NTOK +
                 ((h2 & 7) << 3) + (w2 & 7);
    float v[6];
    float sum = 0.f;
#pragma unroll
    for (int i = 0; i < 6; ++i) { v[i] = MM[mmrow * CC + lane + i * 32]; sum += v[i]; }
    for (int d = 16; d >= 1; d >>= 1) sum += __shfl_xor(sum, d, 32);
    float mean = sum * (1.0f / CC);
    float var = 0.f;
#pragma unroll
    for (int i = 0; i < 6; ++i) { float dv = v[i] - mean; var += dv * dv; }
    for (int d = 16; d >= 1; d >>= 1) var += __shfl_xor(var, d, 32);
    float inv = rsqrtf(var * (1.0f / CC) + 1e-5f);
#pragma unroll
    for (int i = 0; i < 6; ++i) {
        int c = lane + i * 32;
        float sc = x[(((long)b * CC + c) << 16) + (h << 8) + w];
        float o = (v[i] - mean) * inv * g[c] + bt[c] + sc;
        X1[p * CC + c] = o;
        X1b[p * CC + c] = f2bf(o);
    }
}

// ---------------- final: out = x1 + LN(mlp) --------------------------------
__global__ __launch_bounds__(256) void final_ln_add(
    const float* __restrict__ MLP, const float* __restrict__ X1,
    const float* __restrict__ g, const float* __restrict__ bt,
    float* __restrict__ out) {
    const int lane = threadIdx.x & 31;
    const int widx = threadIdx.x >> 5;
    long p = (long)blockIdx.x * 8 + widx;
    if (p >= (long)MTOK) return;
    float v[6];
    float sum = 0.f;
#pragma unroll
    for (int i = 0; i < 6; ++i) { v[i] = MLP[p * CC + lane + i * 32]; sum += v[i]; }
    for (int d = 16; d >= 1; d >>= 1) sum += __shfl_xor(sum, d, 32);
    float mean = sum * (1.0f / CC);
    float var = 0.f;
#pragma unroll
    for (int i = 0; i < 6; ++i) { float dv = v[i] - mean; var += dv * dv; }
    for (int d = 16; d >= 1; d >>= 1) var += __shfl_xor(var, d, 32);
    float inv = rsqrtf(var * (1.0f / CC) + 1e-5f);
#pragma unroll
    for (int i = 0; i < 6; ++i) {
        int c = lane + i * 32;
        out[p * CC + c] = X1[p * CC + c] + (v[i] - mean) * inv * g[c] + bt[c];
    }
}

// ---------------- host-side orchestration ----------------------------------
extern "C" void kernel_launch(void* const* d_in, const int* in_sizes, int n_in,
                              void* d_out, int out_size, void* d_ws, size_t ws_size,
                              hipStream_t stream) {
    const float* x        = (const float*)d_in[0];
    const float* feat     = (const float*)d_in[1];
    const float* attnMask = (const float*)d_in[2];
    const int*   relIdx   = (const int*)d_in[3];
    const float* biasTab  = (const float*)d_in[4];
    const float* q_w      = (const float*)d_in[5];
    const float* q_b      = (const float*)d_in[6];
    const float* kv_w     = (const float*)d_in[7];
    const float* kv_b     = (const float*)d_in[8];
    const float* proj_w   = (const float*)d_in[9];
    const float* proj_b   = (const float*)d_in[10];
    const float* merge_w  = (const float*)d_in[11];
    const float* n1g      = (const float*)d_in[12];
    const float* n1b      = (const float*)d_in[13];
    const float* n2g      = (const float*)d_in[14];
    const float* n2b      = (const float*)d_in[15];
    const float* fc1_w    = (const float*)d_in[16];
    const float* fc1_b    = (const float*)d_in[17];
    const float* fc2_w    = (const float*)d_in[18];
    const float* fc2_b    = (const float*)d_in[19];

    char* ws = (char*)d_ws;
    const size_t MB = 1ull << 20;
    // bf16 weights (<1MB total)
    unsigned short* wq     = (unsigned short*)(ws + 0);
    unsigned short* wkv    = (unsigned short*)(ws + 73728);
    unsigned short* wproj  = (unsigned short*)(ws + 221184);
    unsigned short* wmerge = (unsigned short*)(ws + 294912);
    unsigned short* wfc1   = (unsigned short*)(ws + 368640);
    unsigned short* wfc2   = (unsigned short*)(ws + 663552);
    // big buffers with lifetime-based reuse
    unsigned short* Xw  = (unsigned short*)(ws + 1 * MB);    // 96MB; later O, later X1b
    unsigned short* Fw  = (unsigned short*)(ws + 97 * MB);   // 96MB; later P
    unsigned short* Qb  = (unsigned short*)(ws + 193 * MB);  // 96MB
    unsigned short* KVb = (unsigned short*)(ws + 289 * MB);  // 192MB; later MM f32, later MLP f32
    float*          X1  = (float*)(ws + 481 * MB);           // 192MB, long-lived
    unsigned short* H1  = (unsigned short*)(ws + 673 * MB);  // 384MB
    unsigned short* Ob   = Xw;                // attention out reuses Xw
    unsigned short* Pb   = Fw;                // proj out reuses Fw
    float*          MM   = (float*)KVb;       // merge out reuses KV region
    unsigned short* X1b  = Xw;                // bf16 copy of x1 reuses Xw/O
    float*          MLP  = (float*)KVb;       // fc2 out reuses MM region

    const float scale = 0.17677669529663687f;  // 32^-0.5

    // 1) weights -> bf16
    cvt_bf16<<<(36864 + 255) / 256, 256, 0, stream>>>(q_w, wq, 36864);
    cvt_bf16<<<(73728 + 255) / 256, 256, 0, stream>>>(kv_w, wkv, 73728);
    cvt_bf16<<<(36864 + 255) / 256, 256, 0, stream>>>(proj_w, wproj, 36864);
    cvt_bf16<<<(36864 + 255) / 256, 256, 0, stream>>>(merge_w, wmerge, 36864);
    cvt_bf16<<<(147456 + 255) / 256, 256, 0, stream>>>(fc1_w, wfc1, 147456);
    cvt_bf16<<<(147456 + 255) / 256, 256, 0, stream>>>(fc2_w, wfc2, 147456);

    // 2) shifted-window gather
    {
        long tot = (long)BB * CC * HH * WW2;
        gather_windows<<<(int)((tot + 255) / 256), 256, 0, stream>>>(x, feat, Xw, Fw);
    }

    // 3) Q = (Xw q_w^T + q_b) * scale   [M x 192]
    gemm_bf16<0, 0><<<(MTOK / 16) * (192 / 64) / 8, 256, 0, stream>>>(
        Xw, wq, q_b, Qb, MTOK, 192, 192, scale);
    // 4) KV = Fw kv_w^T + kv_b          [M x 384]
    gemm_bf16<0, 0><<<(MTOK / 16) * (384 / 64) / 8, 256, 0, stream>>>(
        Fw, wkv, kv_b, KVb, MTOK, 192, 384, 1.0f);

    // 5) windowed attention -> Ob [M x 192]
    attn_kernel<<<(BB * NWIN * NHD) / 8, 256, 0, stream>>>(
        Qb, KVb, attnMask, relIdx, biasTab, Ob);

    // 6) proj: P = Ob proj_w^T + proj_b [M x 192] bf16
    gemm_bf16<0, 0><<<(MTOK / 16) * (192 / 64) / 8, 256, 0, stream>>>(
        Ob, wproj, proj_b, Pb, MTOK, 192, 192, 1.0f);
    // 7) merge: MM = P merge_w^T        [M x 192] f32
    gemm_bf16<0, 1><<<(MTOK / 16) * (192 / 64) / 8, 256, 0, stream>>>(
        Pb, wmerge, nullptr, MM, MTOK, 192, 192, 1.0f);

    // 8) x1 = shortcut + LN(MM)  (window order -> NHWC), bf16 copy for MLP
    merge_ln_add<<<MTOK / 8, 256, 0, stream>>>(MM, x, n1g, n1b, X1, X1b);

    // 9) H1 = gelu(X1b fc1_w^T + fc1_b) [M x 768] bf16
    gemm_bf16<1, 0><<<(MTOK / 16) * (768 / 64) / 8, 256, 0, stream>>>(
        X1b, wfc1, fc1_b, H1, MTOK, 192, 768, 1.0f);
    // 10) MLP = H1 fc2_w^T + fc2_b      [M x 192] f32
    gemm_bf16<0, 1><<<(MTOK / 16) * (192 / 64) / 8, 256, 0, stream>>>(
        H1, wfc2, fc2_b, MLP, MTOK, 768, 192, 1.0f);

    // 11) out = x1 + LN(MLP)
    final_ln_add<<<MTOK / 8, 256, 0, stream>>>(MLP, X1, n2g, n2b, (float*)d_out);

    (void)in_sizes; (void)n_in; (void)out_size; (void)ws_size;
}
